// PointNetDecoder_31885837206054
// MI455X (gfx1250) — matrix operand order repaired and checked by reading, not compiled, not run on hardware
//
#include <hip/hip_runtime.h>
#include <hip/hip_bf16.h>

typedef __attribute__((ext_vector_type(16))) _Float16 v16h;
typedef __attribute__((ext_vector_type(8)))  float    v8f;

#define BIGD   1e8f
#define EPS_W  1e-8f
#define EPS_BN 1e-5f
#define KNN_TILE 256
#define GEMM_MT  4      // 16-row tiles per wave (register blocking)

// -------- CDNA5 async global->LDS support (gfx1250), with safe fallback -----
#if defined(__has_builtin)
# if __has_builtin(__builtin_amdgcn_global_load_async_to_lds_b32)
#  define HAVE_ASYNC_LDS 1
# endif
#endif
#ifndef HAVE_ASYNC_LDS
# define HAVE_ASYNC_LDS 0
#endif

#define AS1 __attribute__((address_space(1)))
#define AS3 __attribute__((address_space(3)))

__device__ __forceinline__ void async_b32(const void* g, void* l)
{
#if HAVE_ASYNC_LDS
    // builtin signature (probed): (global int*, shared int*, imm offset, imm cpol)
    __builtin_amdgcn_global_load_async_to_lds_b32(
        (AS1 int*)(void*)g, (AS3 int*)l, 0, 0);
#else
    *(volatile unsigned*)l = *(const unsigned*)g;
#endif
}

__device__ __forceinline__ void wait_async_lds()
{
#if HAVE_ASYNC_LDS
# if defined(__has_builtin) && __has_builtin(__builtin_amdgcn_s_wait_asynccnt)
    __builtin_amdgcn_s_wait_asynccnt(0);
# else
    asm volatile("s_wait_asynccnt 0x0" ::: "memory");
# endif
#endif
}

// ---------------------------------------------------------------------------
// Kernel 1: piece-masked 3-NN over coarse points.
// Double-buffered coarse tiles staged in LDS via async global->LDS copies;
// one thread per fine point maintains a running best-3 (squared distances).
// ---------------------------------------------------------------------------
__global__ void knn3_kernel(const float* __restrict__ fxyz,   // [3][N]
                            const float* __restrict__ cxyz,   // [3][S]
                            const int*   __restrict__ fid,    // [N]
                            const int*   __restrict__ cid,    // [S]
                            int*   __restrict__ out_idx,      // [N][3]
                            float* __restrict__ out_w,        // [N][3]
                            int N, int S)
{
    __shared__ float scx[2][KNN_TILE];
    __shared__ float scy[2][KNN_TILE];
    __shared__ float scz[2][KNN_TILE];
    __shared__ int   sid[2][KNN_TILE];

    const int i = blockIdx.x * blockDim.x + threadIdx.x;
    const int t = threadIdx.x;

    float fx = 0.f, fy = 0.f, fz = 0.f;
    int   pid = -1;
    if (i < N) {
        fx  = fxyz[i];
        fy  = fxyz[N + i];
        fz  = fxyz[2 * N + i];
        pid = fid[i];
    }

    float d0 = BIGD, d1 = BIGD, d2v = BIGD;
    int   i0 = 0,   i1 = 0,   i2 = 0;

    // prologue: stage tile 0 into buffer 0
    if (t < S) {
        async_b32(&cxyz[t],         &scx[0][t]);
        async_b32(&cxyz[S + t],     &scy[0][t]);
        async_b32(&cxyz[2 * S + t], &scz[0][t]);
        async_b32(&cid[t],          &sid[0][t]);
    }

    int buf = 0;
    for (int base = 0; base < S; base += KNN_TILE) {
        const int cnt = min(KNN_TILE, S - base);

        wait_async_lds();     // this wave's in-flight tile copies done
        __syncthreads();      // whole block's tile visible in LDS

        // overlap: start staging the next tile into the other buffer
        const int nbase = base + KNN_TILE;
        if (nbase < S && t < S - nbase) {
            async_b32(&cxyz[nbase + t],         &scx[buf ^ 1][t]);
            async_b32(&cxyz[S + nbase + t],     &scy[buf ^ 1][t]);
            async_b32(&cxyz[2 * S + nbase + t], &scz[buf ^ 1][t]);
            async_b32(&cid[nbase + t],          &sid[buf ^ 1][t]);
        }

        for (int j = 0; j < cnt; ++j) {
            const float dx = fx - scx[buf][j];
            const float dy = fy - scy[buf][j];
            const float dz = fz - scz[buf][j];
            float d = dx * dx + dy * dy + dz * dz;
            d = (sid[buf][j] == pid) ? d : BIGD;
            const int cj = base + j;
            if (d < d0)       { d2v = d1; i2 = i1; d1 = d0; i1 = i0; d0 = d; i0 = cj; }
            else if (d < d1)  { d2v = d1; i2 = i1; d1 = d;  i1 = cj; }
            else if (d < d2v) { d2v = d;  i2 = cj; }
        }
        buf ^= 1;
    }

    if (i < N) {
        const float w0 = 1.f / (d0  + EPS_W);
        const float w1 = 1.f / (d1  + EPS_W);
        const float w2 = 1.f / (d2v + EPS_W);
        const float inv = 1.f / (w0 + w1 + w2);
        out_idx[i * 3 + 0] = i0;
        out_idx[i * 3 + 1] = i1;
        out_idx[i * 3 + 2] = i2;
        out_w[i * 3 + 0] = w0 * inv;
        out_w[i * 3 + 1] = w1 * inv;
        out_w[i * 3 + 2] = w2 * inv;
    }
}

// ---------------------------------------------------------------------------
// Kernel 2: interpolate coarse features + concat fine features -> f16 X [N][K1]
// ---------------------------------------------------------------------------
__global__ void interp_concat_kernel(const float* __restrict__ CF,  // [Dc][S]
                                     const float* __restrict__ FF,  // [Df][N]
                                     const int*   __restrict__ idx, // [N][3]
                                     const float* __restrict__ w,   // [N][3]
                                     _Float16* __restrict__ Xh,     // [N][Dc+Df]
                                     int N, int S, int Dc, int Df)
{
    const int i = blockIdx.x;
    const int c = threadIdx.x;
    const int K = Dc + Df;
    if (i >= N || c >= K) return;
    float v;
    if (c < Dc) {
        const int   j0 = idx[i * 3 + 0], j1 = idx[i * 3 + 1], j2 = idx[i * 3 + 2];
        const float w0 = w[i * 3 + 0],  w1 = w[i * 3 + 1],  w2 = w[i * 3 + 2];
        const float* cfr = CF + (size_t)c * S;
        v = w0 * cfr[j0] + w1 * cfr[j1] + w2 * cfr[j2];
    } else {
        v = FF[(size_t)(c - Dc) * N + i];
    }
    Xh[(size_t)i * K + c] = (_Float16)v;
}

// ---------------------------------------------------------------------------
// f32 -> f16 conversion (weights)
// ---------------------------------------------------------------------------
__global__ void f32_to_f16_kernel(const float* __restrict__ src,
                                  _Float16* __restrict__ dst, int n)
{
    const int t = blockIdx.x * blockDim.x + threadIdx.x;
    if (t < n) dst[t] = (_Float16)src[t];
}

// ---------------------------------------------------------------------------
// Kernel 3/5: WMMA GEMM  Y[M][Cout] = A[M][K] (f16) * W[Cout][K]^T (f16) + bias
// One wave32 per (64-row x 16-col) task: 4 register-blocked 16x16 tiles share
// each B fragment, K consumed in chunks of 32 via v_wmma_f32_16x16x32_f16.
// Fragment K order: contiguous k = 16*half + i, same for A and B (wave32).
// ---------------------------------------------------------------------------
__global__ void wmma_gemm_f16f32_kernel(const _Float16* __restrict__ A,   // [M][K]
                                        const _Float16* __restrict__ W,   // [Cout][K]
                                        const float*    __restrict__ bias,// [Cout]
                                        float*          __restrict__ Y,   // [M][Cout]
                                        int M, int K, int Cout, int totalTasks)
{
    const int wave = threadIdx.x >> 5;            // 8 waves per block
    const int lane = threadIdx.x & 31;
    const int task = blockIdx.x * 8 + wave;
    if (task >= totalTasks) return;               // wave-uniform exit (EXEC stays full)

    const int tilesN = Cout >> 4;
    const int tM0 = (task / tilesN) * GEMM_MT;    // first 16-row tile index
    const int tN  = task % tilesN;

    const int half = lane >> 4;                   // 0: lanes 0-15, 1: lanes 16-31
    const int m    = lane & 15;
    const int col  = tN * 16 + m;

    const _Float16* wcol = W + (size_t)col * K;   // B column (= W row)

    const _Float16* arow[GEMM_MT];
#pragma unroll
    for (int tt = 0; tt < GEMM_MT; ++tt) {
        int r = (tM0 + tt) * 16 + m;
        if (r >= M) r = M - 1;                    // tail clamp (results discarded)
        arow[tt] = A + (size_t)r * K;
    }

    v8f acc[GEMM_MT] = {};

    for (int k0 = 0; k0 < K; k0 += 32) {
        __builtin_prefetch(wcol + k0 + 128, 0, 0);   // global_prefetch_b8
        // B 32x16 f16: lane n=m, halves hold K = k0 + 16*half + i (contiguous)
        v16h b;
#pragma unroll
        for (int i = 0; i < 16; ++i) b[i] = wcol[k0 + (half << 4) + i];

#pragma unroll
        for (int tt = 0; tt < GEMM_MT; ++tt) {
            // A 16x32 f16: lane m, halves hold K = k0 + 16*half + i (contiguous)
            v16h a;
#pragma unroll
            for (int i = 0; i < 16; ++i) {
                const int ka = k0 + (half << 4) + i;
                a[i] = arow[tt][ka];
            }
            acc[tt] = __builtin_amdgcn_wmma_f32_16x16x32_f16(
                          /*neg_a=*/false, a, /*neg_b=*/false, b,
                          /*c_mod=*/(short)0, acc[tt],
                          /*reuse_a=*/false, /*reuse_b=*/false);
        }
    }

    // C/D layout: VGPR r holds (M = r + 8*half, N = lane&15)
    const float bv = bias[col];
#pragma unroll
    for (int tt = 0; tt < GEMM_MT; ++tt) {
        const int rbase = (tM0 + tt) * 16;
        if (rbase >= M) break;
        float* yout = Y + (size_t)rbase * Cout + col;
#pragma unroll
        for (int r = 0; r < 8; ++r)
            yout[(size_t)(r + 8 * half) * Cout] = acc[tt][r] + bv;
    }
}

// ---------------------------------------------------------------------------
// Kernel 4/6: per-channel batch stats -> scale/shift (training-mode BN)
// ---------------------------------------------------------------------------
__global__ void bn_stats_kernel(const float* __restrict__ Y,      // [N][C]
                                const float* __restrict__ gamma,
                                const float* __restrict__ beta,
                                float* __restrict__ scale,
                                float* __restrict__ shift,
                                int N, int C)
{
    __shared__ float ssum[256];
    __shared__ float ssq[256];
    const int ch = blockIdx.x;
    const int t  = threadIdx.x;
    float s = 0.f, q = 0.f;
    for (int n = t; n < N; n += blockDim.x) {
        const float v = Y[(size_t)n * C + ch];
        s += v;
        q += v * v;
    }
    ssum[t] = s;
    ssq[t]  = q;
    __syncthreads();
    for (int off = 128; off > 0; off >>= 1) {
        if (t < off) {
            ssum[t] += ssum[t + off];
            ssq[t]  += ssq[t + off];
        }
        __syncthreads();
    }
    if (t == 0) {
        const float invN = 1.f / (float)N;
        const float mu   = ssum[0] * invN;
        const float var  = ssq[0] * invN - mu * mu;
        const float a    = gamma[ch] * rsqrtf(var + EPS_BN);
        scale[ch] = a;
        shift[ch] = beta[ch] - mu * a;
    }
}

// BN+ReLU apply, f16 output (feeds next GEMM), layout unchanged [N][C]
__global__ void bn_relu_f16_kernel(const float* __restrict__ Y,
                                   const float* __restrict__ scale,
                                   const float* __restrict__ shift,
                                   _Float16* __restrict__ Out,
                                   int total, int C)
{
    const int t = blockIdx.x * blockDim.x + threadIdx.x;
    if (t >= total) return;
    const int c = t % C;
    float v = Y[t] * scale[c] + shift[c];
    v = v > 0.f ? v : 0.f;
    Out[t] = (_Float16)v;
}

// BN+ReLU apply + transpose: Out[c][n] = relu(bn(Y[n][c]))
__global__ void bn_relu_transpose_kernel(const float* __restrict__ Y,   // [N][C]
                                         const float* __restrict__ scale,
                                         const float* __restrict__ shift,
                                         float* __restrict__ Out,       // [C][N]
                                         int N, int C)
{
    const int t = blockIdx.x * blockDim.x + threadIdx.x;
    if (t >= N * C) return;
    const int n = t / C;
    const int c = t % C;
    float v = Y[t] * scale[c] + shift[c];
    v = v > 0.f ? v : 0.f;
    Out[(size_t)c * N + n] = v;
}

// ---------------------------------------------------------------------------
extern "C" void kernel_launch(void* const* d_in, const int* in_sizes, int n_in,
                              void* d_out, int out_size, void* d_ws, size_t ws_size,
                              hipStream_t stream)
{
    const float* fxyz = (const float*)d_in[0];
    const float* cxyz = (const float*)d_in[1];
    const int*   fid  = (const int*)d_in[2];
    const int*   cid  = (const int*)d_in[3];
    const float* FF   = (const float*)d_in[4];
    const float* CF   = (const float*)d_in[5];
    const float* W1   = (const float*)d_in[6];
    const float* b1   = (const float*)d_in[7];
    const float* g1   = (const float*)d_in[8];
    const float* be1  = (const float*)d_in[9];
    const float* W2   = (const float*)d_in[10];
    const float* b2   = (const float*)d_in[11];
    const float* g2   = (const float*)d_in[12];
    const float* be2  = (const float*)d_in[13];

    const int N  = in_sizes[0] / 3;     // 16384
    const int S  = in_sizes[1] / 3;     // 4096
    const int Df = in_sizes[4] / N;     // 64
    const int Dc = in_sizes[5] / S;     // 256
    const int C1 = in_sizes[7];         // 256
    const int C2 = in_sizes[11];        // 128
    const int K1 = Dc + Df;             // 320
    const int K2 = C1;                  // 256

    // --- workspace layout ---
    char* ws = (char*)d_ws;
    auto alloc = [&](size_t bytes) -> char* {
        char* p = ws;
        ws += (bytes + 255) & ~(size_t)255;
        return p;
    };
    int*      idx    = (int*)     alloc((size_t)N * 3 * sizeof(int));
    float*    w      = (float*)   alloc((size_t)N * 3 * sizeof(float));
    _Float16* Xh     = (_Float16*)alloc((size_t)N * K1 * sizeof(_Float16));
    _Float16* W1h    = (_Float16*)alloc((size_t)C1 * K1 * sizeof(_Float16));
    _Float16* W2h    = (_Float16*)alloc((size_t)C2 * K2 * sizeof(_Float16));
    float*    Y1     = (float*)   alloc((size_t)N * C1 * sizeof(float));
    _Float16* Y1h    = (_Float16*)alloc((size_t)N * C1 * sizeof(_Float16));
    float*    Y2     = (float*)   alloc((size_t)N * C2 * sizeof(float));
    float*    scale1 = (float*)   alloc((size_t)C1 * sizeof(float));
    float*    shift1 = (float*)   alloc((size_t)C1 * sizeof(float));
    float*    scale2 = (float*)   alloc((size_t)C2 * sizeof(float));
    float*    shift2 = (float*)   alloc((size_t)C2 * sizeof(float));
    (void)ws_size;

    // 1) 3-NN search (async-LDS double buffered)
    knn3_kernel<<<dim3((N + KNN_TILE - 1) / KNN_TILE), dim3(KNN_TILE), 0, stream>>>(
        fxyz, cxyz, fid, cid, idx, w, N, S);

    // 2) interpolate + concat -> f16 activations
    interp_concat_kernel<<<dim3(N), dim3(K1), 0, stream>>>(
        CF, FF, idx, w, Xh, N, S, Dc, Df);

    // weight conversions
    {
        const int n1 = C1 * K1, n2 = C2 * K2;
        f32_to_f16_kernel<<<dim3((n1 + 255) / 256), dim3(256), 0, stream>>>(W1, W1h, n1);
        f32_to_f16_kernel<<<dim3((n2 + 255) / 256), dim3(256), 0, stream>>>(W2, W2h, n2);
    }

    // 3) GEMM1: [N][K1] x [C1][K1]^T -> [N][C1]
    {
        const int tasks = ((N + 16 * GEMM_MT - 1) / (16 * GEMM_MT)) * (C1 / 16);
        wmma_gemm_f16f32_kernel<<<dim3((tasks + 7) / 8), dim3(256), 0, stream>>>(
            Xh, W1h, b1, Y1, N, K1, C1, tasks);
    }

    // 4) BN stats + apply (layer 1) -> f16
    bn_stats_kernel<<<dim3(C1), dim3(256), 0, stream>>>(Y1, g1, be1, scale1, shift1, N, C1);
    bn_relu_f16_kernel<<<dim3((N * C1 + 255) / 256), dim3(256), 0, stream>>>(
        Y1, scale1, shift1, Y1h, N * C1, C1);

    // 5) GEMM2: [N][C1] x [C2][C1]^T -> [N][C2]
    {
        const int tasks = ((N + 16 * GEMM_MT - 1) / (16 * GEMM_MT)) * (C2 / 16);
        wmma_gemm_f16f32_kernel<<<dim3((tasks + 7) / 8), dim3(256), 0, stream>>>(
            Y1h, W2h, b2, Y2, N, K2, C2, tasks);
    }

    // 6) BN stats + apply + transpose -> out [C2][N]
    bn_stats_kernel<<<dim3(C2), dim3(256), 0, stream>>>(Y2, g2, be2, scale2, shift2, N, C2);
    bn_relu_transpose_kernel<<<dim3((N * C2 + 255) / 256), dim3(256), 0, stream>>>(
        Y2, scale2, shift2, (float*)d_out, N, C2);
}
